// MySCNN_87222195847213
// MI455X (gfx1250) — matrix operand order repaired and checked by reading, not compiled, not run on hardware
//
#include <hip/hip_runtime.h>

// ---------------------------------------------------------------------------
// SCNN forward for gfx1250 (MI455X).
// Layout: node-major feature maps F[(m*2+b) * 128 + c] (f32, ld=128).
// Dense channel mixing: v_wmma_f32_16x16x32_bf16; theta staged to LDS via
// global_load_async_to_lds_b128 (ASYNCcnt) once per workgroup; B operands
// stream from LDS (ds_load_b128); A cvt'd f32->bf16 with v_cvt_pk_bf16_f32.
// SpMM: float4 coalesced gathers + global_atomic_add_f32 scatters.
// Bias + leaky-ReLU fused into the last GEMM epilogue.
// ---------------------------------------------------------------------------

typedef __attribute__((ext_vector_type(16))) __bf16 v16bf;
typedef __attribute__((ext_vector_type(8)))  float  v8f;

#define LDF 128  // fixed leading dimension (floats) of all feature buffers

__device__ __forceinline__ void atomicAddF32(float* p, float v) {
  // lowers to global_atomic_add_f32 (no return) on gfx1250
  __hip_atomic_fetch_add(p, v, __ATOMIC_RELAXED, __HIP_MEMORY_SCOPE_AGENT);
}

// ---- input transpose: x (B=2, C=4, M) -> X[(m*2+b)*LDF + c], pad c to 32 ----
__global__ void scnn_tin(float* __restrict__ X, const float* __restrict__ x, int M) {
  int tid = blockIdx.x * blockDim.x + threadIdx.x;
  int total = 2 * M * 32;
  if (tid >= total) return;
  int r = tid >> 5;        // row = m*2 + b
  int c = tid & 31;
  int m = r >> 1;
  int b = r & 1;
  float v = (c < 4) ? x[((size_t)b * 4 + c) * M + m] : 0.0f;
  X[(size_t)r * LDF + c] = v;
}

// ---- theta prep: th (Cout, Cin, 5) f32 -> 5 padded bf16 mats [Nd x Kd] ----
__global__ void scnn_ptheta(__bf16* __restrict__ dst, const float* __restrict__ th,
                            int Cout, int Cin, int Nd, int Kd) {
  int tid = blockIdx.x * blockDim.x + threadIdx.x;
  int nk = Nd * Kd;
  if (tid >= 5 * nk) return;
  int k = tid / nk;
  int rem = tid - k * nk;
  int o = rem / Kd;
  int i = rem - o * Kd;
  float v = (o < Cout && i < Cin) ? th[((size_t)o * Cin + i) * 5 + k] : 0.0f;
  dst[tid] = (__bf16)v;   // dst[k*nk + o*Kd + i]
}

// ---- zero the active Kd columns of a buffer, float4-wide ----
__global__ void scnn_zero4(float* __restrict__ dst, int total, int gBits4, int gMask4) {
  int tid = blockIdx.x * blockDim.x + threadIdx.x;
  if (tid >= total) return;
  int r = tid >> gBits4;
  int g = tid & gMask4;
  float4 z; z.x = 0.f; z.y = 0.f; z.z = 0.f; z.w = 0.f;
  *(float4*)(dst + (size_t)r * LDF + (g << 2)) = z;
}

// ---- SpMM scatter: dst[row,:] += val * src[col,:] over both batch rows ----
// Q = Kd/2 lanes per nnz; each lane owns one float4 channel group of one batch
__global__ void scnn_spmm(float* __restrict__ dst, const float* __restrict__ src,
                          const int* __restrict__ rows, const int* __restrict__ cols,
                          const float* __restrict__ vals, int nnz,
                          int qShift, int gBits, int gMask) {
  int tid = blockIdx.x * blockDim.x + threadIdx.x;
  int e = tid >> qShift;
  if (e >= nnz) return;
  int j  = tid & ((1 << qShift) - 1);
  int b  = j >> gBits;          // batch 0/1
  int ch = (j & gMask) << 2;    // channel group start
  int r  = rows[e];
  int c  = cols[e];
  float v = vals[e];
  float4 s = *(const float4*)(src + ((size_t)c * 2 + b) * LDF + ch);
  float* d = dst + ((size_t)r * 2 + b) * LDF + ch;
  atomicAddF32(d + 0, v * s.x);
  atomicAddF32(d + 1, v * s.y);
  atomicAddF32(d + 2, v * s.z);
  atomicAddF32(d + 3, v * s.w);
}

// ---- WMMA GEMM row-strip kernel -------------------------------------------
// Block = 128 threads = 4 waves; wave w computes the 16-row strip
// rowBase = (blockIdx.x*4 + w)*16 across all NT column tiles.
// theta_k [Nd x Kd] is staged into LDS once per block with async copies.
template <int KT, int NT, bool ACC>
__global__ __launch_bounds__(128)
void scnn_gemm_wmma(float* __restrict__ Y, const float* __restrict__ X,
                    const __bf16* __restrict__ T,
                    const float* __restrict__ bias, int Cout, int relu) {
  constexpr int Kd = KT * 32;
  constexpr int NELEM = NT * 16 * Kd;          // theta elems (bf16)
  __shared__ __align__(32) __bf16 shT[NELEM];

  // ---- async-stage theta into LDS (16B chunks, ASYNCcnt tracked) ----
  {
    const int chunks = NELEM / 8;              // 16 bytes per chunk
    for (int i = threadIdx.x; i < chunks; i += 128) {   // chunks % 128 == 0
      unsigned lds = (unsigned)(uintptr_t)(&shT[0]) + (unsigned)i * 16u;
      unsigned long long ga =
          (unsigned long long)(uintptr_t)T + (unsigned long long)i * 16ull;
      asm volatile("global_load_async_to_lds_b128 %0, %1, off"
                   :: "v"(lds), "v"(ga) : "memory");
    }
    asm volatile("s_wait_asynccnt 0x0" ::: "memory");
  }
  __syncthreads();

  const int lane = threadIdx.x & 31;
  const int wave = threadIdx.x >> 5;
  const int half = lane >> 4;
  const int l15  = lane & 15;
  const int rowBase = (blockIdx.x * 4 + wave) * 16;

  v8f c[NT];
#pragma unroll
  for (int n = 0; n < NT; ++n) {
    if (ACC) {
#pragma unroll
      for (int r = 0; r < 8; ++r)
        c[n][r] = Y[(size_t)(rowBase + r + half * 8) * LDF + n * 16 + l15];
    } else {
#pragma unroll
      for (int r = 0; r < 8; ++r) c[n][r] = 0.0f;
    }
  }

  // A: lane holds row M=l15, 16 contiguous K values kb..kb+15 (kb = half*16)
  const float* ap = X + (size_t)(rowBase + l15) * LDF + half * 16;
  // B (from LDS): lane holds column N (output channel), 16 contiguous K values
  const __bf16* bp = &shT[0] + (size_t)l15 * Kd + half * 16;

#pragma unroll
  for (int kt = 0; kt < KT; ++kt) {
    v16bf a;
    const float* a0 = ap + kt * 32;
#pragma unroll
    for (int e = 0; e < 8; ++e) {
      a[e]     = (__bf16)a0[e];
      a[e + 8] = (__bf16)a0[e + 8];
    }
#pragma unroll
    for (int n = 0; n < NT; ++n) {
      v16bf b = *(const v16bf*)(bp + (size_t)(n * 16) * Kd + kt * 32);
      c[n] = __builtin_amdgcn_wmma_f32_16x16x32_bf16(false, a, false, b,
                                                     (short)0, c[n], false, false);
    }
  }

#pragma unroll
  for (int n = 0; n < NT; ++n) {
    const int col = n * 16 + l15;
    float bv = (bias != nullptr && col < Cout) ? bias[col] : 0.0f;
#pragma unroll
    for (int r = 0; r < 8; ++r) {
      float v = c[n][r] + bv;
      if (relu) v = v > 0.0f ? v : 0.01f * v;
      Y[(size_t)(rowBase + r + half * 8) * LDF + col] = v;
    }
  }
}

// ---- output transpose: Y[(m*2+b)*LDF + c] -> out[b*4*M + c*M + m] ----
__global__ void scnn_tout(float* __restrict__ out, const float* __restrict__ Y, int M) {
  int tid = blockIdx.x * blockDim.x + threadIdx.x;
  int total = 8 * M;
  if (tid >= total) return;
  int b   = tid / (4 * M);
  int rem = tid - b * 4 * M;
  int cch = rem / M;
  int m   = rem - cch * M;
  out[tid] = Y[((size_t)m * 2 + b) * LDF + cch];
}

static inline int cdiv(long long a, long long b) { return (int)((a + b - 1) / b); }

extern "C" void kernel_launch(void* const* d_in, const int* in_sizes, int n_in,
                              void* d_out, int out_size, void* d_ws, size_t ws_size,
                              hipStream_t stream) {
  (void)in_sizes; (void)n_in; (void)out_size; (void)ws_size;

  static const int MS[3]  = {40000, 100000, 60000};
  static const int NNZ[3] = {400000, 1000000, 600000};
  // per-layer geometry (channel pads): Kd = padded Cin, Nd = padded Cout
  static const int KD[3]  = {32, 128, 128};
  static const int ND[3]  = {128, 128, 16};
  static const int CIN[3] = {4, 120, 120};
  static const int COUT[3]= {120, 120, 4};

  // workspace: three ping-pong feature buffers (max R=200000 rows x 128 f32)
  const size_t P_ELEMS = (size_t)200000 * LDF;
  float* P0 = (float*)d_ws;
  float* P1 = P0 + P_ELEMS;
  float* P2 = P1 + P_ELEMS;
  __bf16* TH = (__bf16*)(P2 + P_ELEMS);   // 5 * 128 * 128 bf16 max

  float* outF = (float*)d_out;
  size_t outOff = 0;

  for (int t = 0; t < 3; ++t) {
    const int M   = MS[t];
    const int nnz = NNZ[t];
    const int R   = 2 * M;                       // rows (m*2 + b)
    const float* x  = (const float*)d_in[4 * t + 0];
    const int*   rw = (const int*)  d_in[4 * t + 1];
    const int*   cl = (const int*)  d_in[4 * t + 2];
    const float* vl = (const float*)d_in[4 * t + 3];

    float* X  = P0;   // current layer input
    float* Yb = P1;   // current layer output accumulator
    float* S  = P2;   // SpMM scratch

    scnn_tin<<<cdiv((long long)R * 32, 256), 256, 0, stream>>>(X, x, M);

    for (int j = 0; j < 3; ++j) {
      const int Kd = KD[j], Nd = ND[j], Cin = CIN[j], Cout = COUT[j];
      const float* th = (const float*)d_in[12 + 6 * t + 2 * j];
      const float* bi = (const float*)d_in[13 + 6 * t + 2 * j];
      const int relu = (j < 2) ? 1 : 0;

      // prep 5 padded bf16 theta matrices [Nd x Kd]
      scnn_ptheta<<<cdiv((long long)5 * Nd * Kd, 256), 256, 0, stream>>>(
          TH, th, Cout, Cin, Nd, Kd);

      const dim3 gg(R / 64);                     // 4 row tiles per block
      const int nk = Nd * Kd;

      // gemm dispatcher: per-layer (KT, NT) shape, ACC flag, fused epilogue
      auto gemm = [&](bool acc, float* Yp, const float* Xp, const __bf16* Tk,
                      const float* bp, int rl) {
        if (j == 0) {
          if (acc) scnn_gemm_wmma<1, 8, true ><<<gg, 128, 0, stream>>>(Yp, Xp, Tk, bp, Cout, rl);
          else     scnn_gemm_wmma<1, 8, false><<<gg, 128, 0, stream>>>(Yp, Xp, Tk, bp, Cout, rl);
        } else if (j == 1) {
          if (acc) scnn_gemm_wmma<4, 8, true ><<<gg, 128, 0, stream>>>(Yp, Xp, Tk, bp, Cout, rl);
          else     scnn_gemm_wmma<4, 8, false><<<gg, 128, 0, stream>>>(Yp, Xp, Tk, bp, Cout, rl);
        } else {
          if (acc) scnn_gemm_wmma<4, 1, true ><<<gg, 128, 0, stream>>>(Yp, Xp, Tk, bp, Cout, rl);
          else     scnn_gemm_wmma<4, 1, false><<<gg, 128, 0, stream>>>(Yp, Xp, Tk, bp, Cout, rl);
        }
      };

      // k = 0 term (writes Y; no epilogue yet)
      gemm(false, Yb, X, TH, nullptr, 0);

      // k = 1..4: scratch = L * cur ; Y += scratch * theta_k
      const int kdBits = (Kd == 128) ? 7 : 5;
      const int gBits  = kdBits - 2;        // log2(Kd/4)
      const int gMask  = (Kd >> 2) - 1;
      const int qShift = kdBits - 1;        // log2(2*Kd/4) lanes per nnz
      float* cur = X;
      float* nxt = S;
      for (int k = 1; k < 5; ++k) {
        scnn_zero4<<<cdiv((long long)R * (Kd >> 2), 256), 256, 0, stream>>>(
            nxt, R * (Kd >> 2), gBits, gMask);
        scnn_spmm<<<cdiv(((long long)nnz) << qShift, 256), 256, 0, stream>>>(
            nxt, cur, rw, cl, vl, nnz, qShift, gBits, gMask);
        // fuse bias + leaky-ReLU into the final accumulating GEMM
        const float* bp = (k == 4) ? bi : nullptr;
        const int    rl = (k == 4) ? relu : 0;
        gemm(true, Yb, nxt, TH + (size_t)k * nk, bp, rl);
        float* tmp = cur; cur = nxt; nxt = tmp;
      }

      if (j < 2) {
        // rotate buffers: output becomes next input; old input is reused
        float* tmp = X; X = Yb; Yb = tmp;   // S keeps serving as scratch
      } else {
        scnn_tout<<<cdiv((long long)8 * M, 256), 256, 0, stream>>>(outF + outOff, Yb, M);
      }
    }
    outOff += (size_t)8 * M;
  }
}
